// SpeakerEmbeddingPredictor_41772851921012
// MI455X (gfx1250) — matrix operand order repaired and checked by reading, not compiled, not run on hardware
//
#include <hip/hip_runtime.h>
#include <hip/hip_bf16.h>
#include <math.h>

// ---------------------------------------------------------------------------
// Conformer forward for MI455X (gfx1250, wave32, WMMA bf16 f32-accumulate)
// ---------------------------------------------------------------------------

typedef __attribute__((ext_vector_type(16))) __bf16 v16bf;
typedef __attribute__((ext_vector_type(8)))  __bf16 v8bf;
typedef __attribute__((ext_vector_type(8)))  float  v8f;

#define DI static __device__ __forceinline__

#define Bn   8
#define Tn   1024
#define Dn   256
#define Hn   8
#define DKn  32
#define Ln   5
#define FFn  2048
#define Kc   31
#define MBT  (Bn * Tn)     // 8192
#define NPE  (2 * Tn - 1)  // 2047
#define NPEP 2048          // padded rows for pe/pp (row 2047 zeroed)

// Async global->LDS staging (gfx1250) if the toolchain exposes it.
#if defined(__has_builtin)
#if __has_builtin(__builtin_amdgcn_global_load_async_to_lds_b128)
#define HAVE_ASYNC_LDS 1
#endif
#endif

#if defined(HAVE_ASYNC_LDS)
typedef int ai4 __attribute__((vector_size(16)));
#define ASYNC_CP16(dst, src)                                                  \
  __builtin_amdgcn_global_load_async_to_lds_b128(                             \
      (__attribute__((address_space(1))) ai4*)(src),                          \
      (__attribute__((address_space(3))) ai4*)(dst), 0, 0)
DI void async_wait0() {
#if __has_builtin(__builtin_amdgcn_s_wait_asynccnt)
  __builtin_amdgcn_s_wait_asynccnt(0);
#else
  asm volatile("s_wait_asynccnt 0" ::: "memory");
#endif
}
#endif

DI float sigmoidf_(float x) { return 1.0f / (1.0f + __expf(-x)); }

DI v8f wmma_bf16(v16bf a, v16bf b, v8f c) {
  return __builtin_amdgcn_wmma_f32_16x16x32_bf16(
      /*neg_a=*/false, a, /*neg_b=*/false, b,
      /*c_mod=*/(short)0, c, /*reuse_a=*/false, /*reuse_b=*/false);
}

// A fragment (16x32 bf16) from f32 row-major data (global or LDS).
// a[e] holds A[m = lane&15][k = 16*(lane>>4) + e]
DI v16bf frag_a_f32(const float* rowp, int hh) {
  v16bf a;
#pragma unroll
  for (int s2 = 0; s2 < 2; ++s2) {
    int kb = 16 * hh + 8 * s2;
    float4 f0 = *(const float4*)(rowp + kb);
    float4 f1 = *(const float4*)(rowp + kb + 4);
    a[8 * s2 + 0] = (__bf16)f0.x; a[8 * s2 + 1] = (__bf16)f0.y;
    a[8 * s2 + 2] = (__bf16)f0.z; a[8 * s2 + 3] = (__bf16)f0.w;
    a[8 * s2 + 4] = (__bf16)f1.x; a[8 * s2 + 5] = (__bf16)f1.y;
    a[8 * s2 + 6] = (__bf16)f1.z; a[8 * s2 + 7] = (__bf16)f1.w;
  }
  return a;
}

// A fragment with additive bias row (q + pos_u/pos_v)
DI v16bf frag_a_add(const float* rowp, const float* addp, int hh) {
  v16bf a;
#pragma unroll
  for (int s2 = 0; s2 < 2; ++s2) {
    int kb = 16 * hh + 8 * s2;
    float4 f0 = *(const float4*)(rowp + kb);
    float4 f1 = *(const float4*)(rowp + kb + 4);
    float4 a0 = *(const float4*)(addp + kb);
    float4 a1 = *(const float4*)(addp + kb + 4);
    a[8 * s2 + 0] = (__bf16)(f0.x + a0.x); a[8 * s2 + 1] = (__bf16)(f0.y + a0.y);
    a[8 * s2 + 2] = (__bf16)(f0.z + a0.z); a[8 * s2 + 3] = (__bf16)(f0.w + a0.w);
    a[8 * s2 + 4] = (__bf16)(f1.x + a1.x); a[8 * s2 + 5] = (__bf16)(f1.y + a1.y);
    a[8 * s2 + 6] = (__bf16)(f1.z + a1.z); a[8 * s2 + 7] = (__bf16)(f1.w + a1.w);
  }
  return a;
}

// A fragment from bf16 row-major matrix (attention probabilities)
DI v16bf frag_a_bf(const __bf16* rowp, int hh) {
  v16bf a;
#pragma unroll
  for (int s2 = 0; s2 < 2; ++s2) {
    int kb = 16 * hh + 8 * s2;
    v8bf t = *(const v8bf*)(rowp + kb);
#pragma unroll
    for (int j = 0; j < 8; ++j) a[8 * s2 + j] = t[j];
  }
  return a;
}

// B fragment (32x16 bf16): b[e] holds B[k = 16*(lane>>4) + e][n = lane&15];
// p must point at W_row(n) + k0 + 16*(lane>>4): 16 contiguous f32.
DI v16bf frag_b16(const float* p) {
  v16bf b;
#pragma unroll
  for (int qd = 0; qd < 4; ++qd) {
    float4 f = ((const float4*)p)[qd];
    b[4 * qd + 0] = (__bf16)f.x; b[4 * qd + 1] = (__bf16)f.y;
    b[4 * qd + 2] = (__bf16)f.z; b[4 * qd + 3] = (__bf16)f.w;
  }
  return b;
}

DI float block_sum(float* red, int tid) {
  __syncthreads();
#pragma unroll
  for (int o = 128; o > 0; o >>= 1) {
    if (tid < o) red[tid] += red[tid + o];
    __syncthreads();
  }
  float r = red[0];
  __syncthreads();
  return r;
}

// ---------------------------------------------------------------------------
// GEMM: C[M,N] = act[M,K] @ W[N,K]^T (+bias) (+swish) (+residual)
// LDS-staged: block tile 32(M) x 128(N) x 64(K); 256 threads = 8 waves,
// each wave computes a 16x32 tile (2 accumulators, A fragment reused).
// Requires M % 32 == 0, N % 128 == 0, K % 64 == 0.
// ---------------------------------------------------------------------------
__global__ void k_gemm(const float* __restrict__ A, const float* __restrict__ W,
                       const float* __restrict__ bias, const float* __restrict__ res,
                       float* __restrict__ C, int M, int N, int K, int epi) {
  __shared__ float Af[32][68];
  __shared__ float Wf[128][68];
  int tid = threadIdx.x;
  int lane = tid & 31;
  int w = tid >> 5;
  int wm = w & 1, wn = w >> 1;
  int ln15 = lane & 15, hh = lane >> 4;
  int mblk = blockIdx.y * 32;
  int nblk = blockIdx.x * 128;

  // staging coordinates: A tile 32x64 (8 f32/thread), W tile 128x64 (32 f32/thread)
  int ar = tid >> 3, ak = (tid & 7) * 8;
  int wr = tid >> 1, wk = (tid & 1) * 32;
  const float* Ars = A + (size_t)(mblk + ar) * K + ak;
  const float* Wrs = W + (size_t)(nblk + wr) * K + wk;

  v8f c0 = {0.f, 0.f, 0.f, 0.f, 0.f, 0.f, 0.f, 0.f};
  v8f c1 = {0.f, 0.f, 0.f, 0.f, 0.f, 0.f, 0.f, 0.f};

  for (int k0 = 0; k0 < K; k0 += 64) {
#if defined(HAVE_ASYNC_LDS)
#pragma unroll
    for (int j = 0; j < 2; ++j)
      ASYNC_CP16(&Af[ar][ak + 4 * j], Ars + k0 + 4 * j);
#pragma unroll
    for (int j = 0; j < 8; ++j)
      ASYNC_CP16(&Wf[wr][wk + 4 * j], Wrs + k0 + 4 * j);
    async_wait0();
#else
#pragma unroll
    for (int j = 0; j < 2; ++j)
      *(float4*)&Af[ar][ak + 4 * j] = *(const float4*)(Ars + k0 + 4 * j);
#pragma unroll
    for (int j = 0; j < 8; ++j)
      *(float4*)&Wf[wr][wk + 4 * j] = *(const float4*)(Wrs + k0 + 4 * j);
#endif
    __syncthreads();

#pragma unroll
    for (int kk2 = 0; kk2 < 64; kk2 += 32) {
      v16bf a = frag_a_f32(&Af[wm * 16 + ln15][kk2], hh);
      v16bf b0 = frag_b16(&Wf[wn * 32 + ln15][kk2 + 16 * hh]);
      v16bf b1 = frag_b16(&Wf[wn * 32 + 16 + ln15][kk2 + 16 * hh]);
      c0 = wmma_bf16(a, b0, c0);
      c1 = wmma_bf16(a, b1, c1);
    }
    __syncthreads();
  }

  int col0 = nblk + wn * 32 + ln15;
  float bv0 = bias ? bias[col0] : 0.0f;
  float bv1 = bias ? bias[col0 + 16] : 0.0f;
#pragma unroll
  for (int r = 0; r < 8; ++r) {
    int row = mblk + r + 8 * hh;
    float v0 = c0[r] + bv0;
    float v1 = c1[r] + bv1;
    if (epi == 1) { v0 = v0 * sigmoidf_(v0); v1 = v1 * sigmoidf_(v1); }
    if (res) {
      v0 += res[(size_t)row * N + col0];
      v1 += res[(size_t)row * N + col0 + 16];
    }
    C[(size_t)row * N + col0] = v0;
    C[(size_t)row * N + col0 + 16] = v1;
  }
}

// ---------------------------------------------------------------------------
// Attention scores for one batch: sc[h,t,s] = (qu.k + qv.p[s-t+1023]) / sqrt(32)
// grid (T/16, H, 8), block 256 (8 waves, wave handles one s-tile)
// ---------------------------------------------------------------------------
__global__ void k_scores(const float* __restrict__ q, const float* __restrict__ kk,
                         const float* __restrict__ pp, const float* __restrict__ pu,
                         const float* __restrict__ pv, float* __restrict__ sc) {
  __shared__ float Gs[8][16][32];
  int lane = threadIdx.x & 31;
  int w = threadIdx.x >> 5;
  int ln15 = lane & 15, hh = lane >> 4;
  int head = blockIdx.y;
  int t0 = blockIdx.x * 16;
  int s0 = (blockIdx.z * 8 + w) * 16;

  const float* qrow = q + (size_t)(t0 + ln15) * Dn + head * DKn;
  v16bf aqu = frag_a_add(qrow, pu + head * DKn, hh);
  v16bf aqv = frag_a_add(qrow, pv + head * DKn, hh);

  v16bf bk = frag_b16(kk + (size_t)(s0 + ln15) * Dn + head * DKn + 16 * hh);
  v8f zero = {0.f, 0.f, 0.f, 0.f, 0.f, 0.f, 0.f, 0.f};
  v8f ac = wmma_bf16(aqu, bk, zero);

  // G band: two 16x16 WMMAs against p rows [j0, j0+32)
  int j0 = s0 - t0 + 1008;
  int r1 = j0 + ln15;      r1 = r1 < 0 ? 0 : (r1 > NPE - 1 ? NPE - 1 : r1);
  int r2 = j0 + 16 + ln15; r2 = r2 < 0 ? 0 : (r2 > NPE - 1 ? NPE - 1 : r2);
  v16bf bp1 = frag_b16(pp + (size_t)r1 * Dn + head * DKn + 16 * hh);
  v16bf bp2 = frag_b16(pp + (size_t)r2 * Dn + head * DKn + 16 * hh);
  v8f g1 = wmma_bf16(aqv, bp1, zero);
  v8f g2 = wmma_bf16(aqv, bp2, zero);

#pragma unroll
  for (int r = 0; r < 8; ++r) {
    Gs[w][r + 8 * hh][ln15] = g1[r];
    Gs[w][r + 8 * hh][16 + ln15] = g2[r];
  }
  __syncthreads();

  const float scl = 0.17677669529663687f;  // 1/sqrt(32)
#pragma unroll
  for (int r = 0; r < 8; ++r) {
    int tt = r + 8 * hh;
    int ss = ln15;
    float bd = Gs[w][tt][ss - tt + 15];  // j = (s0+ss)-(t0+tt)+1023 in band
    sc[((size_t)head * Tn + (t0 + tt)) * Tn + s0 + ss] = (ac[r] + bd) * scl;
  }
}

// ---------------------------------------------------------------------------
// Row softmax with mask, f32 -> bf16 probabilities. grid H*T, block 256.
// ---------------------------------------------------------------------------
__global__ void k_softmax(const float* __restrict__ sc, const unsigned char* __restrict__ mask,
                          __bf16* __restrict__ pbf) {
  __shared__ float red[256];
  int rid = blockIdx.x;
  int tid = threadIdx.x;
  const float* row = sc + (size_t)rid * Tn;
  __bf16* prow = pbf + (size_t)rid * Tn;

  float v[4];
  unsigned char mk[4];
  float mx = -3.0e38f;
#pragma unroll
  for (int i = 0; i < 4; ++i) {
    int s = tid + i * 256;
    mk[i] = mask[s];
    v[i] = mk[i] ? row[s] : -1.0e30f;
    mx = fmaxf(mx, v[i]);
  }
  red[tid] = mx;
  __syncthreads();
#pragma unroll
  for (int o = 128; o > 0; o >>= 1) {
    if (tid < o) red[tid] = fmaxf(red[tid], red[tid + o]);
    __syncthreads();
  }
  float m = red[0];
  __syncthreads();

  float e[4];
  float s0 = 0.f;
#pragma unroll
  for (int i = 0; i < 4; ++i) { e[i] = __expf(v[i] - m); s0 += e[i]; }
  red[tid] = s0;
  float tot = block_sum(red, tid);
  float inv = 1.0f / tot;
#pragma unroll
  for (int i = 0; i < 4; ++i) {
    int s = tid + i * 256;
    prow[s] = (__bf16)(mk[i] ? e[i] * inv : 0.0f);
  }
}

// ---------------------------------------------------------------------------
// out[t, h*32+dk] = sum_s P[h,t,s] * V[s, h*32+dk]; grid (T/16, H), block 64.
// ---------------------------------------------------------------------------
__global__ void k_pv(const __bf16* __restrict__ pbf, const float* __restrict__ vv,
                     float* __restrict__ ao) {
  int lane = threadIdx.x & 31;
  int w = threadIdx.x >> 5;  // 0..1 -> dk halves
  int ln15 = lane & 15, hh = lane >> 4;
  int head = blockIdx.y;
  int t0 = blockIdx.x * 16;
  int n0 = w * 16;

  const __bf16* prow = pbf + ((size_t)head * Tn + t0 + ln15) * Tn;
  v8f c = {0.f, 0.f, 0.f, 0.f, 0.f, 0.f, 0.f, 0.f};
  for (int k0 = 0; k0 < Tn; k0 += 32) {
    __builtin_prefetch(prow + k0 + 64, 0, 3);
    v16bf a = frag_a_bf(prow + k0, hh);
    v16bf b;
#pragma unroll
    for (int e = 0; e < 16; ++e)
      b[e] = (__bf16)vv[(size_t)(k0 + 16 * hh + e) * Dn + head * DKn + n0 + ln15];
    c = wmma_bf16(a, b, c);
  }
#pragma unroll
  for (int r = 0; r < 8; ++r)
    ao[(size_t)(t0 + r + 8 * hh) * Dn + head * DKn + n0 + ln15] = c[r];
}

// ---------------------------------------------------------------------------
// LayerNorm over last dim (256). grid = rows, block 256. Safe in-place.
// ---------------------------------------------------------------------------
__global__ void k_ln(const float* __restrict__ x, const float* __restrict__ g,
                     const float* __restrict__ b, float* __restrict__ y) {
  __shared__ float red[256];
  int row = blockIdx.x, d = threadIdx.x;
  float v = x[(size_t)row * Dn + d];
  red[d] = v;
  float mu = block_sum(red, d) * (1.0f / Dn);
  float dv = v - mu;
  red[d] = dv * dv;
  float var = block_sum(red, d) * (1.0f / Dn);
  y[(size_t)row * Dn + d] = dv * rsqrtf(var + 1e-12f) * g[d] + b[d];
}

// Embedding: x = LN(xs @ Wemb^T + bemb) * sqrt(D). grid rows, block 256.
__global__ void k_embed(const float* __restrict__ xs, const float* __restrict__ Wemb,
                        const float* __restrict__ bemb, const float* __restrict__ g,
                        const float* __restrict__ b, float* __restrict__ x) {
  __shared__ float sx[80];
  __shared__ float red[256];
  int row = blockIdx.x, d = threadIdx.x;
  if (d < 80) sx[d] = xs[(size_t)row * 80 + d];
  __syncthreads();
  float acc = bemb[d];
  const float* wr = Wemb + (size_t)d * 80;
  for (int j = 0; j < 80; ++j) acc += sx[j] * wr[j];
  red[d] = acc;
  float mu = block_sum(red, d) * (1.0f / Dn);
  float dv = acc - mu;
  red[d] = dv * dv;
  float var = block_sum(red, d) * (1.0f / Dn);
  x[(size_t)row * Dn + d] = (dv * rsqrtf(var + 1e-12f) * g[d] + b[d]) * 16.0f;
}

// Relative positional embedding table pe[2048, 256]; row 2047 zeroed (padding)
__global__ void k_posemb(float* __restrict__ pe) {
  int i = blockIdx.x * 256 + threadIdx.x;
  if (i >= NPEP * Dn) return;
  int n = i >> 8, d = i & 255;
  if (n >= NPE) { pe[i] = 0.0f; return; }
  int j = d >> 1;
  float div = __expf(-logf(10000.0f) * (float)(2 * j) / (float)Dn);
  float p = (n < Tn) ? (float)(Tn - 1 - n) : -(float)(n - Tn + 1);
  float a = p * div;
  pe[i] = (d & 1) ? cosf(a) : sinf(a);
}

// GLU: yg = y1[:, :256] * sigmoid(y1[:, 256:512])
__global__ void k_glu(const float* __restrict__ y1, float* __restrict__ yg) {
  int i = blockIdx.x * 256 + threadIdx.x;
  int row = i >> 8, c = i & 255;
  float a = y1[(size_t)row * 512 + c];
  float g = y1[(size_t)row * 512 + 256 + c];
  yg[i] = a * sigmoidf_(g);
}

// Depthwise conv along T, K=31, zero pad 15
__global__ void k_dwconv(const float* __restrict__ yg, const float* __restrict__ wgt,
                         const float* __restrict__ bias, float* __restrict__ z) {
  int i = blockIdx.x * 256 + threadIdx.x;
  int c = i & 255;
  int t = (i >> 8) & (Tn - 1);
  int bb = i >> 18;
  float acc = bias[c];
  const float* wc = wgt + c * Kc;
#pragma unroll
  for (int k2 = 0; k2 < Kc; ++k2) {
    int tt = t + k2 - 15;
    if (tt >= 0 && tt < Tn) acc += yg[((size_t)bb * Tn + tt) * Dn + c] * wc[k2];
  }
  z[i] = acc;
}

// BatchNorm statistics per channel over (B,T)
__global__ void k_bnstats(const float* __restrict__ z, float* __restrict__ mu,
                          float* __restrict__ inv) {
  __shared__ float r1[256], r2[256];
  int c = blockIdx.x, tid = threadIdx.x;
  float s = 0.f, ss = 0.f;
  for (int i = tid; i < MBT; i += 256) {
    float v = z[(size_t)i * Dn + c];
    s += v;
    ss += v * v;
  }
  r1[tid] = s;
  r2[tid] = ss;
  __syncthreads();
#pragma unroll
  for (int o = 128; o > 0; o >>= 1) {
    if (tid < o) { r1[tid] += r1[tid + o]; r2[tid] += r2[tid + o]; }
    __syncthreads();
  }
  if (tid == 0) {
    float m = r1[0] * (1.0f / MBT);
    float v = r2[0] * (1.0f / MBT) - m * m;
    mu[c] = m;
    inv[c] = rsqrtf(v + 1e-5f);
  }
}

// BN normalize + swish
__global__ void k_bnswish(const float* __restrict__ z, const float* __restrict__ mu,
                          const float* __restrict__ inv, const float* __restrict__ g,
                          const float* __restrict__ b, float* __restrict__ y2) {
  int i = blockIdx.x * 256 + threadIdx.x;
  int c = i & 255;
  float v = (z[i] - mu[c]) * inv[c] * g[c] + b[c];
  y2[i] = v * sigmoidf_(v);
}

// Final sum over T: out[b,d] = sum_t x[b,t,d]
__global__ void k_sumT(const float* __restrict__ x, float* __restrict__ out) {
  int b = blockIdx.x, d = threadIdx.x;
  float s = 0.f;
  const float* p = x + (size_t)b * Tn * Dn + d;
  for (int t = 0; t < Tn; ++t) s += p[(size_t)t * Dn];
  out[b * Dn + d] = s;
}

// ---------------------------------------------------------------------------
// Host launcher
// ---------------------------------------------------------------------------
extern "C" void kernel_launch(void* const* d_in, const int* in_sizes, int n_in,
                              void* d_out, int out_size, void* d_ws, size_t ws_size,
                              hipStream_t stream) {
  (void)in_sizes; (void)n_in; (void)out_size; (void)ws_size;

  const float* xs      = (const float*)d_in[0];
  const float* Wemb    = (const float*)d_in[1];
  const float* bemb    = (const float*)d_in[2];
  const float* ln_in_g = (const float*)d_in[3];
  const float* ln_in_b = (const float*)d_in[4];
  const float* Wq = (const float*)d_in[5];
  const float* bq = (const float*)d_in[6];
  const float* Wk = (const float*)d_in[7];
  const float* bk = (const float*)d_in[8];
  const float* Wv = (const float*)d_in[9];
  const float* bv = (const float*)d_in[10];
  const float* Wo = (const float*)d_in[11];
  const float* bo = (const float*)d_in[12];
  const float* Wp = (const float*)d_in[13];
  const float* pos_u = (const float*)d_in[14];
  const float* pos_v = (const float*)d_in[15];
  const float* ln1_g = (const float*)d_in[16];
  const float* ln1_b = (const float*)d_in[17];
  const float* lnc_g = (const float*)d_in[18];
  const float* lnc_b = (const float*)d_in[19];
  const float* lnf_g = (const float*)d_in[20];
  const float* lnf_b = (const float*)d_in[21];
  const float* lnfin_g = (const float*)d_in[22];
  const float* lnfin_b = (const float*)d_in[23];
  const float* pw1_w = (const float*)d_in[24];
  const float* pw1_b = (const float*)d_in[25];
  const float* dw_w  = (const float*)d_in[26];
  const float* dw_b  = (const float*)d_in[27];
  const float* bn_g  = (const float*)d_in[28];
  const float* bn_b  = (const float*)d_in[29];
  const float* pw2_w = (const float*)d_in[30];
  const float* pw2_b = (const float*)d_in[31];
  const float* ff1_w = (const float*)d_in[32];
  const float* ff1_b = (const float*)d_in[33];
  const float* ff2_w = (const float*)d_in[34];
  const float* ff2_b = (const float*)d_in[35];
  const float* after_g = (const float*)d_in[36];
  const float* after_b = (const float*)d_in[37];
  const unsigned char* mask = (const unsigned char*)d_in[38];

  float* out = (float*)d_out;
  float* ws = (float*)d_ws;

  // workspace layout (float offsets)
  float* x    = ws;
  float* h    = ws + 2097152ull;
  float* q    = ws + 4194304ull;
  float* kk   = ws + 6291456ull;
  float* vv   = ws + 8388608ull;
  float* ao   = ws + 10485760ull;
  float* y1   = ws + 12582912ull;   // (B,T,512)
  float* yg   = ws + 16777216ull;
  float* z    = ws + 18874368ull;
  float* y2   = ws + 20971520ull;
  float* pe   = ws + 23068672ull;   // (2048,256), row 2047 zero
  float* pp   = ws + 23592960ull;   // (2048,256), row 2047 zero
  float* bnmu = ws + 24117248ull;
  float* bninv= ws + 24117504ull;
  float* h1   = ws + 25165824ull;   // (B,T,2048) -- aliased by attention scratch
  float* scores = h1;                            // (H,T,T) f32, per batch
  __bf16* pbf = (__bf16*)(ws + 33554432ull);     // (H,T,T) bf16, per batch

  dim3 b256(256);

  k_posemb<<<dim3(NPEP * Dn / 256), b256, 0, stream>>>(pe);
  k_embed<<<dim3(MBT), b256, 0, stream>>>(xs, Wemb, bemb, ln_in_g, ln_in_b, x);

  for (int i = 0; i < Ln; ++i) {
    const float* Wq_i = Wq + (size_t)i * Dn * Dn;
    const float* Wk_i = Wk + (size_t)i * Dn * Dn;
    const float* Wv_i = Wv + (size_t)i * Dn * Dn;
    const float* Wo_i = Wo + (size_t)i * Dn * Dn;
    const float* Wp_i = Wp + (size_t)i * Dn * Dn;
    const float* bq_i = bq + (size_t)i * Dn;
    const float* bk_i = bk + (size_t)i * Dn;
    const float* bv_i = bv + (size_t)i * Dn;
    const float* bo_i = bo + (size_t)i * Dn;
    const float* pu_i = pos_u + (size_t)i * Hn * DKn;
    const float* pv_i = pos_v + (size_t)i * Hn * DKn;

    // ---- attention ----
    k_ln<<<dim3(MBT), b256, 0, stream>>>(x, ln1_g + i * Dn, ln1_b + i * Dn, h);
    k_gemm<<<dim3(Dn / 128, MBT / 32), b256, 0, stream>>>(h, Wq_i, bq_i, nullptr, q, MBT, Dn, Dn, 0);
    k_gemm<<<dim3(Dn / 128, MBT / 32), b256, 0, stream>>>(h, Wk_i, bk_i, nullptr, kk, MBT, Dn, Dn, 0);
    k_gemm<<<dim3(Dn / 128, MBT / 32), b256, 0, stream>>>(h, Wv_i, bv_i, nullptr, vv, MBT, Dn, Dn, 0);
    k_gemm<<<dim3(Dn / 128, NPEP / 32), b256, 0, stream>>>(pe, Wp_i, nullptr, nullptr, pp, NPEP, Dn, Dn, 0);

    for (int bb = 0; bb < Bn; ++bb) {
      const float* qb = q + (size_t)bb * Tn * Dn;
      const float* kb2 = kk + (size_t)bb * Tn * Dn;
      const float* vb = vv + (size_t)bb * Tn * Dn;
      float* aob = ao + (size_t)bb * Tn * Dn;
      k_scores<<<dim3(Tn / 16, Hn, 8), b256, 0, stream>>>(qb, kb2, pp, pu_i, pv_i, scores);
      k_softmax<<<dim3(Hn * Tn), b256, 0, stream>>>(scores, mask + (size_t)bb * Tn, pbf);
      k_pv<<<dim3(Tn / 16, Hn), dim3(64), 0, stream>>>(pbf, vb, aob);
    }
    k_gemm<<<dim3(Dn / 128, MBT / 32), b256, 0, stream>>>(ao, Wo_i, bo_i, x, x, MBT, Dn, Dn, 0);

    // ---- conv module ----
    k_ln<<<dim3(MBT), b256, 0, stream>>>(x, lnc_g + i * Dn, lnc_b + i * Dn, h);
    k_gemm<<<dim3(512 / 128, MBT / 32), b256, 0, stream>>>(h, pw1_w + (size_t)i * 512 * Dn,
                                                           pw1_b + (size_t)i * 512, nullptr, y1,
                                                           MBT, 512, Dn, 0);
    k_glu<<<dim3(MBT * Dn / 256), b256, 0, stream>>>(y1, yg);
    k_dwconv<<<dim3(MBT * Dn / 256), b256, 0, stream>>>(yg, dw_w + (size_t)i * Dn * Kc,
                                                        dw_b + (size_t)i * Dn, z);
    k_bnstats<<<dim3(Dn), b256, 0, stream>>>(z, bnmu, bninv);
    k_bnswish<<<dim3(MBT * Dn / 256), b256, 0, stream>>>(z, bnmu, bninv, bn_g + i * Dn,
                                                         bn_b + i * Dn, y2);
    k_gemm<<<dim3(Dn / 128, MBT / 32), b256, 0, stream>>>(y2, pw2_w + (size_t)i * Dn * Dn,
                                                          pw2_b + (size_t)i * Dn, x, x,
                                                          MBT, Dn, Dn, 0);

    // ---- feed-forward ----
    k_ln<<<dim3(MBT), b256, 0, stream>>>(x, lnf_g + i * Dn, lnf_b + i * Dn, h);
    k_gemm<<<dim3(FFn / 128, MBT / 32), b256, 0, stream>>>(h, ff1_w + (size_t)i * FFn * Dn,
                                                           ff1_b + (size_t)i * FFn, nullptr, h1,
                                                           MBT, FFn, Dn, 1 /*swish*/);
    k_gemm<<<dim3(Dn / 128, MBT / 32), b256, 0, stream>>>(h1, ff2_w + (size_t)i * FFn * Dn,
                                                          ff2_b + (size_t)i * Dn, x, x,
                                                          MBT, Dn, FFn, 0);
    k_ln<<<dim3(MBT), b256, 0, stream>>>(x, lnfin_g + i * Dn, lnfin_b + i * Dn, x);
  }

  k_ln<<<dim3(MBT), b256, 0, stream>>>(x, after_g, after_b, x);
  k_sumT<<<dim3(Bn), b256, 0, stream>>>(x, out);
}
